// CapsNet_86079734547131
// MI455X (gfx1250) — matrix operand order repaired and checked by reading, not compiled, not run on hardware
//
#include <hip/hip_runtime.h>
#include <hip/hip_bf16.h>
#include <math.h>

typedef __attribute__((ext_vector_type(16))) __bf16 v16bf;
typedef __attribute__((ext_vector_type(8)))  float  v8f;
typedef __attribute__((ext_vector_type(2)))  float  v2f;

// ---------------------------------------------------------------------------
// Kernel 1: conv1 (1 in-ch, 9x9, stride 1) + bias + ReLU -> h NHWC bf16
// h[b][oy][ox][oc], oy,ox in 0..19, oc in 0..255
// ---------------------------------------------------------------------------
__global__ __launch_bounds__(256) void k_conv1(const float* __restrict__ x,
                                               const float* __restrict__ w,
                                               const float* __restrict__ bias,
                                               __bf16* __restrict__ h) {
  const int pos = blockIdx.x;            // 0 .. 256*20*20-1
  const int ox = pos % 20;
  const int t1 = pos / 20;
  const int oy = t1 % 20;
  const int b  = t1 / 20;
  __shared__ float xs[81];
  const int tid = threadIdx.x;
  if (tid < 81) {
    const int ky = tid / 9, kx = tid % 9;
    xs[tid] = x[b * 784 + (oy + ky) * 28 + (ox + kx)];
  }
  __syncthreads();
  const int oc = tid;
  const float* wp = w + oc * 81;
  float acc = bias[oc];
#pragma unroll
  for (int p = 0; p < 81; ++p) acc = fmaf(xs[p], wp[p], acc);
  acc = acc > 0.0f ? acc : 0.0f;
  h[(size_t)pos * 256 + oc] = (__bf16)acc;
}

// ---------------------------------------------------------------------------
// Kernel 2: repack pri_w [8,32,256,9,9] fp32 -> wBt[n][k] bf16
// n = l*32+o (256), k = (ky*9+kx)*256 + c (20736)
// ---------------------------------------------------------------------------
__global__ __launch_bounds__(256) void k_repack(const float* __restrict__ pw,
                                                __bf16* __restrict__ wBt) {
  const size_t g = (size_t)blockIdx.x * 256 + threadIdx.x;  // 256*20736 total
  const int n = (int)(g / 20736);
  const int k = (int)(g % 20736);
  const int pix = k >> 8;
  const int c = k & 255;
  wBt[g] = (__bf16)pw[((size_t)n * 256 + c) * 81 + pix];
}

// ---------------------------------------------------------------------------
// Kernel 3: primary-caps conv as implicit GEMM, bf16 WMMA, f32 accumulate.
// M = 9216 (b*36 + oy*6 + ox), N = 256 (l*32+o), K = 20736.
// Block tile 128x64, 8 waves (4 M-waves x 2 N-waves), wave tile 32x32.
// 64-K stage (324 stages), register-prefetch software pipeline:
//   barrier; LDS<-regs; barrier; issue global loads for s+1; 8 WMMA on stage s.
// A 64-K stage never crosses a pixel boundary (256 % 64 == 0), so per-thread
// global loads are fully contiguous (64B for A, 32B for B).
// ---------------------------------------------------------------------------
__global__ __launch_bounds__(256) void k_capsconv(const __bf16* __restrict__ h,
                                                  const __bf16* __restrict__ wBt,
                                                  float* __restrict__ out) {
  __shared__ __bf16 As[128 * 72];   // [row 0..127][k 0..63], row stride 72
  __shared__ __bf16 Bs[64 * 72];    // [n 0..63][k 0..63], row stride 72

  const int tid  = threadIdx.x;
  const int lane = tid & 31;
  const int wave = tid >> 5;
  const int mw = wave & 3;          // 0..3
  const int nw = wave >> 2;         // 0..1
  const int Mb = blockIdx.x % 72;
  const int Nb = blockIdx.x / 72;   // 0..3

  // A staging: thread -> (row, 32-elem k-half); 4x uint4 contiguous
  const int ar = tid >> 1;              // 0..127
  const int ac = (tid & 1) * 32;        // 0 / 32
  const int m  = Mb * 128 + ar;
  const int bidx = m / 36;
  const int s0   = m % 36;
  const int oy = s0 / 6, ox = s0 % 6;
  // B staging: thread -> (n, 16-elem k-quarter); 2x uint4 contiguous
  const int bn = tid & 63;
  const int bc = (tid >> 6) * 16;       // 0,16,32,48
  const size_t wrow = (size_t)(Nb * 64 + bn) * 20736;

  v8f acc[2][2];
#pragma unroll
  for (int i = 0; i < 2; ++i)
#pragma unroll
    for (int j = 0; j < 2; ++j)
      acc[i][j] = (v8f){0.f, 0.f, 0.f, 0.f, 0.f, 0.f, 0.f, 0.f};

  const int hi = lane >> 4;             // 0/1
  const int ln = lane & 15;

  union Frag { uint4 u[2]; v16bf v; };

  // ---- prefetch stage 0 into registers ----
  uint4 apf0, apf1, apf2, apf3, bpf0, bpf1;
  {
    const __bf16* ap = h + (((size_t)bidx * 20 + 2 * oy) * 20 + 2 * ox) * 256 + ac;
    const uint4* ap4 = reinterpret_cast<const uint4*>(ap);
    apf0 = ap4[0]; apf1 = ap4[1]; apf2 = ap4[2]; apf3 = ap4[3];
    const uint4* bp4 = reinterpret_cast<const uint4*>(wBt + wrow + bc);
    bpf0 = bp4[0]; bpf1 = bp4[1];
  }

  for (int s = 0; s < 324; ++s) {
    __syncthreads();
    {
      uint4* d = reinterpret_cast<uint4*>(&As[ar * 72 + ac]);
      d[0] = apf0; d[1] = apf1; d[2] = apf2; d[3] = apf3;
      uint4* e = reinterpret_cast<uint4*>(&Bs[bn * 72 + bc]);
      e[0] = bpf0; e[1] = bpf1;
    }
    __syncthreads();

    // ---- issue global loads for stage s+1; they overlap the WMMAs below ----
    if (s + 1 < 324) {
      const int k0  = (s + 1) * 64;
      const int pix = k0 >> 8;          // constant within a stage
      const int c0  = k0 & 255;
      const int ky = pix / 9, kx = pix % 9;
      const __bf16* ap = h + (((size_t)bidx * 20 + (2 * oy + ky)) * 20 +
                              (2 * ox + kx)) * 256 + c0 + ac;
      const uint4* ap4 = reinterpret_cast<const uint4*>(ap);
      apf0 = ap4[0]; apf1 = ap4[1]; apf2 = ap4[2]; apf3 = ap4[3];
      const uint4* bp4 = reinterpret_cast<const uint4*>(wBt + wrow + k0 + bc);
      bpf0 = bp4[0]; bpf1 = bp4[1];
    }

    // ---- compute: 2 k-substeps x (2x2 tiles) = 8 WMMA per wave ----
#pragma unroll
    for (int ks = 0; ks < 2; ++ks) {
      v16bf a[2], bfr[2];
#pragma unroll
      for (int i = 0; i < 2; ++i) {
        const int row = mw * 32 + i * 16 + ln;
        Frag f;
        f.u[0] = *reinterpret_cast<const uint4*>(&As[row * 72 + ks * 32 + hi * 8]);
        f.u[1] = *reinterpret_cast<const uint4*>(&As[row * 72 + ks * 32 + hi * 8 + 16]);
        a[i] = f.v;
      }
#pragma unroll
      for (int j = 0; j < 2; ++j) {
        const int col = nw * 32 + j * 16 + ln;
        Frag f;
        f.u[0] = *reinterpret_cast<const uint4*>(&Bs[col * 72 + ks * 32 + hi * 8]);
        f.u[1] = *reinterpret_cast<const uint4*>(&Bs[col * 72 + ks * 32 + hi * 8 + 16]);
        bfr[j] = f.v;
      }
#pragma unroll
      for (int i = 0; i < 2; ++i)
#pragma unroll
        for (int j = 0; j < 2; ++j)
          acc[i][j] = __builtin_amdgcn_wmma_f32_16x16x32_bf16(
              false, a[i], false, bfr[j], (short)0, acc[i][j], false, false);
    }
  }

#pragma unroll
  for (int i = 0; i < 2; ++i)
#pragma unroll
    for (int j = 0; j < 2; ++j)
#pragma unroll
      for (int v = 0; v < 8; ++v) {
        const int gm = Mb * 128 + mw * 32 + i * 16 + v + 8 * hi;
        const int gn = Nb * 64 + nw * 32 + j * 16 + ln;
        out[(size_t)gm * 256 + gn] = acc[i][j][v];
      }
}

// ---------------------------------------------------------------------------
// Kernel 4: bias + reshape + squash over 1152 -> csq[j][b][m] (transposed for U)
// block = (b, l); caps[b,l,j] = gemm[(b*36 + j%36)][l*32 + j/36] + pri_b[l*32+j/36]
// ---------------------------------------------------------------------------
__global__ __launch_bounds__(256) void k_squash(const float* __restrict__ go,
                                                const float* __restrict__ pb,
                                                float* __restrict__ csq) {
  const int bl = blockIdx.x;            // 0..2047
  const int l = bl & 7;
  const int b = bl >> 3;
  const int tid = threadIdx.x;
  __shared__ float red[256];
  __shared__ float vals[1152];
  float ss = 0.0f;
  for (int j = tid; j < 1152; j += 256) {
    const int o = j / 36, s = j % 36;
    const float v = go[(size_t)(b * 36 + s) * 256 + l * 32 + o] + pb[l * 32 + o];
    vals[j] = v;
    ss += v * v;
  }
  red[tid] = ss;
  __syncthreads();
  for (int st = 128; st > 0; st >>= 1) {
    if (tid < st) red[tid] += red[tid + st];
    __syncthreads();
  }
  const float n2 = red[0];
  const float scale = sqrtf(n2) / (1.0f + n2);   // n/(1+n^2)
  for (int j = tid; j < 1152; j += 256)
    csq[((size_t)j * 256 + b) * 8 + l] = vals[j] * scale;
}

// ---------------------------------------------------------------------------
// Kernel 5: U[b,j,k,l] = sum_m csq[j][b][m] * caps_W[j][m][k*10+l]
// One block per j. f32 WMMA 16x16x4, M=256, N=160, K=8 (2 k-steps).
// ---------------------------------------------------------------------------
__global__ __launch_bounds__(256) void k_topU(const float* __restrict__ csq,
                                              const float* __restrict__ cW,
                                              float* __restrict__ U) {
  const int j = blockIdx.x;             // 0..1151
  __shared__ float As2[256 * 8];
  __shared__ float Bs2[8 * 160];
  const int tid = threadIdx.x;
  {
    const float4* src = reinterpret_cast<const float4*>(csq + (size_t)j * 2048 + tid * 8);
    float4* dst = reinterpret_cast<float4*>(&As2[tid * 8]);
    dst[0] = src[0];
    dst[1] = src[1];
  }
  for (int g = tid; g < 1280; g += 256) Bs2[g] = cW[(size_t)j * 1280 + g];
  __syncthreads();

  const int lane = tid & 31, wave = tid >> 5;
  const int hi = lane >> 4, ln = lane & 15;

  for (int mi = 0; mi < 2; ++mi) {
    const int mt = wave * 2 + mi;       // 16 M-tiles over 8 waves
    for (int nt = 0; nt < 10; ++nt) {
      v8f acc = (v8f){0.f, 0.f, 0.f, 0.f, 0.f, 0.f, 0.f, 0.f};
      const int col = nt * 16 + ln;     // < 160
#pragma unroll
      for (int ks = 0; ks < 2; ++ks) {
        const int row = mt * 16 + ln;
        const int kb = ks * 4 + 2 * hi; // K base for this lane half
        v2f a, bfr;
        a[0] = As2[row * 8 + kb];
        a[1] = As2[row * 8 + kb + 1];
        bfr[0] = Bs2[kb * 160 + col];
        bfr[1] = Bs2[(kb + 1) * 160 + col];
        acc = __builtin_amdgcn_wmma_f32_16x16x4_f32(
            false, a, false, bfr, (short)0, acc, false, false);
      }
#pragma unroll
      for (int v = 0; v < 8; ++v) {
        const int gb = mt * 16 + v + 8 * hi;
        U[((size_t)gb * 1152 + j) * 160 + col] = acc[v];
      }
    }
  }
}

// ---------------------------------------------------------------------------
// Routing kernels
// ---------------------------------------------------------------------------
__global__ void k_zero(float* __restrict__ p, int n) {
  const int i = blockIdx.x * 256 + threadIdx.x;
  if (i < n) p[i] = 0.0f;
}

__global__ __launch_bounds__(256) void k_softmax(const float* __restrict__ bb,
                                                 float* __restrict__ C) {
  const int l = blockIdx.x;             // 0..9
  const int tid = threadIdx.x;
  __shared__ float red[256];
  float mx = -1e30f;
  for (int j = tid; j < 1152; j += 256) mx = fmaxf(mx, bb[l * 1152 + j]);
  red[tid] = mx;
  __syncthreads();
  for (int st = 128; st > 0; st >>= 1) {
    if (tid < st) red[tid] = fmaxf(red[tid], red[tid + st]);
    __syncthreads();
  }
  mx = red[0];
  __syncthreads();
  float sm = 0.0f;
  for (int j = tid; j < 1152; j += 256) sm += __expf(bb[l * 1152 + j] - mx);
  red[tid] = sm;
  __syncthreads();
  for (int st = 128; st > 0; st >>= 1) {
    if (tid < st) red[tid] += red[tid + st];
    __syncthreads();
  }
  const float inv = 1.0f / red[0];
  for (int j = tid; j < 1152; j += 256)
    C[l * 1152 + j] = __expf(bb[l * 1152 + j] - mx) * inv;
}

// S[i,k,l] = sum_j U[i,j,k,l]*C[l,j]; V = squash(S over class dim)
__global__ __launch_bounds__(160) void k_route_sv(const float* __restrict__ U,
                                                  const float* __restrict__ C,
                                                  float* __restrict__ V) {
  const int i = blockIdx.x;             // batch
  const int t = threadIdx.x;            // 0..159 = k*10 + l
  const int l = t % 10;
  __shared__ float S[160];
  float acc = 0.0f;
  const float* Up = U + (size_t)i * 1152 * 160;
  const float* Cl = C + l * 1152;
  for (int j = 0; j < 1152; ++j) acc = fmaf(Up[(size_t)j * 160 + t], Cl[j], acc);
  S[t] = acc;
  __syncthreads();
  const int k = t / 10;
  float n2 = 0.0f;
#pragma unroll
  for (int q = 0; q < 10; ++q) { const float v = S[k * 10 + q]; n2 += v * v; }
  const float scale = sqrtf(n2) / (1.0f + n2);
  V[(size_t)i * 160 + t] = acc * scale;
}

// b[l,j] += sum_{i,k} U[i,j,k,l]*V[i,k,l] / B
__global__ __launch_bounds__(256) void k_route_bu(const float* __restrict__ U,
                                                  const float* __restrict__ V,
                                                  float* __restrict__ bb) {
  const int j = blockIdx.x;             // 0..1151
  const int i = threadIdx.x;            // batch
  float acc[10];
#pragma unroll
  for (int l = 0; l < 10; ++l) acc[l] = 0.0f;
  const float* Up = U + ((size_t)i * 1152 + j) * 160;
  const float* Vp = V + (size_t)i * 160;
#pragma unroll 4
  for (int k = 0; k < 16; ++k)
#pragma unroll
    for (int l = 0; l < 10; ++l)
      acc[l] = fmaf(Up[k * 10 + l], Vp[k * 10 + l], acc[l]);
  __shared__ float red[10][256];
#pragma unroll
  for (int l = 0; l < 10; ++l) red[l][i] = acc[l];
  __syncthreads();
  for (int st = 128; st > 0; st >>= 1) {
    if (i < st)
#pragma unroll
      for (int l = 0; l < 10; ++l) red[l][i] += red[l][i + st];
    __syncthreads();
  }
  if (i < 10) bb[i * 1152 + j] += red[i][0] * (1.0f / 256.0f);
}

// ---------------------------------------------------------------------------
// Host launch
// ---------------------------------------------------------------------------
extern "C" void kernel_launch(void* const* d_in, const int* in_sizes, int n_in,
                              void* d_out, int out_size, void* d_ws, size_t ws_size,
                              hipStream_t stream) {
  (void)in_sizes; (void)n_in; (void)out_size; (void)ws_size;
  const float* x  = (const float*)d_in[0];   // [256,1,28,28]
  const float* w1 = (const float*)d_in[1];   // [256,1,9,9]
  const float* b1 = (const float*)d_in[2];   // [256]
  const float* pw = (const float*)d_in[3];   // [8,32,256,9,9]
  const float* pb = (const float*)d_in[4];   // [8,32]
  const float* cW = (const float*)d_in[5];   // [1152,8,16,10]
  float* out = (float*)d_out;                // [256,16,10]

  char* ws = (char*)d_ws;
  __bf16* h   = (__bf16*)(ws);                    // 52,428,800 B
  __bf16* wBt = (__bf16*)(ws + 52428800);         // 10,616,832 B
  float* gout = (float*)(ws + 63045632);          //  9,437,184 B
  float* csq  = (float*)(ws + 72482816);          //  9,437,184 B
  float* U    = (float*)(ws + 81920000);          // 188,743,680 B
  float* bb   = (float*)(ws + 270663680);         //     46,080 B
  float* Cm   = (float*)(ws + 270709760);         //     46,080 B
  float* Vb   = (float*)(ws + 270755840);         //    163,840 B

  k_conv1<<<256 * 20 * 20, 256, 0, stream>>>(x, w1, b1, h);
  k_repack<<<20736, 256, 0, stream>>>(pw, wBt);
  k_capsconv<<<288, 256, 0, stream>>>(h, wBt, gout);
  k_squash<<<2048, 256, 0, stream>>>(gout, pb, csq);
  k_topU<<<1152, 256, 0, stream>>>(csq, cW, U);
  k_zero<<<(11520 + 255) / 256, 256, 0, stream>>>(bb, 11520);
  for (int r = 0; r < 3; ++r) {
    k_softmax<<<10, 256, 0, stream>>>(bb, Cm);
    k_route_sv<<<256, 160, 0, stream>>>(U, Cm, (r == 2) ? out : Vb);
    if (r < 2) k_route_bu<<<1152, 256, 0, stream>>>(U, Vb, bb);
  }
}
